// FusedSelfCrossMultiHeadDotProductAttention_91207925498088
// MI455X (gfx1250) — matrix operand order repaired and checked by reading, not compiled, not run on hardware
//
#include <hip/hip_runtime.h>
#include <hip/hip_bf16.h>
#include <cstdint>

// B=4, QL=1024, KVS=1024, KVC=256 (KV=1280), D=1024, H=16, HD=64
// All GEMMs have K = N = 1024.

typedef __bf16 bf16;
typedef __attribute__((ext_vector_type(16))) __bf16 bf16x16;
typedef __attribute__((ext_vector_type(8)))  __bf16 bf16x8;
typedef __attribute__((ext_vector_type(4)))  __bf16 bf16x4;
typedef __attribute__((ext_vector_type(8)))  float  f32x8;
typedef __attribute__((ext_vector_type(4)))  unsigned int u32x4;
typedef __attribute__((ext_vector_type(8)))  int i32x8;
typedef __attribute__((ext_vector_type(4)))  int i32x4;

#define DEV __device__ __forceinline__

// ---- WMMA fragment loaders (CDNA5 16-bit layouts, wave32) ------------------
DEV bf16x16 frag_a_ld(const bf16* base, int ld) {
  const int lane = threadIdx.x & 31;
  const int m = lane & 15, h = lane >> 4;
  const bf16* p = base + m * ld + h * 8;
  bf16x8 lo = *(const bf16x8*)(p);
  bf16x8 hi = *(const bf16x8*)(p + 16);
  return __builtin_shufflevector(lo, hi, 0,1,2,3,4,5,6,7,8,9,10,11,12,13,14,15);
}
DEV bf16x16 frag_b_ld(const bf16* base, int ld) {
  const int lane = threadIdx.x & 31;
  const int n = lane & 15, h = lane >> 4;
  const bf16* p = base + n * ld + h * 8;
  bf16x8 lo = *(const bf16x8*)(p);
  bf16x8 hi = *(const bf16x8*)(p + 16);
  return __builtin_shufflevector(lo, hi, 0,1,2,3,4,5,6,7,8,9,10,11,12,13,14,15);
}
DEV f32x8 wmma_bf16(bf16x16 a, bf16x16 b, f32x8 c) {
  return __builtin_amdgcn_wmma_f32_16x16x32_bf16(false, a, false, b, (short)0, c,
                                                 false, false);
}

// ---- generic 4-elem load-as-bf16 and C store -------------------------------
DEV bf16x4 ld4bf(const float* p) {
  float4 v = *(const float4*)p;
  bf16x4 r = {(bf16)v.x, (bf16)v.y, (bf16)v.z, (bf16)v.w};
  return r;
}
DEV bf16x4 ld4bf(const bf16* p) { return *(const bf16x4*)p; }
DEV void st_c(float* p, float v) { *p = v; }
DEV void st_c(bf16* p, float v) { *p = (bf16)v; }

// ---- Tensor Data Mover: 2D bf16 tile -> padded LDS rows --------------------
// cols must be 64 (row = 32 DWORDs); pad_interval=32dw, pad_amount=4dw
// reproduces an LDS pitch of 72 bf16 (144B) in hardware.
DEV void tdm_load_2d_bf16(unsigned lds_off, const bf16* gptr,
                          unsigned rows, unsigned cols, unsigned stride_elems) {
  unsigned long long ga = (unsigned long long)(uintptr_t)gptr;
  u32x4 g0;
  g0[0] = 1u;                                        // count=1, user descriptor
  g0[1] = lds_off;                                   // lds_addr (bytes)
  g0[2] = (unsigned)(ga & 0xffffffffu);              // global_addr[31:0]
  g0[3] = (unsigned)((ga >> 32) & 0x1ffffffu) | (2u << 30);  // addr[56:32]|type=2
  i32x8 g1;
  g1[0] = (int)((1u << 16) |                         // data_size = 2 bytes
                (1u << 20) |                         // pad_enable
                (4u << 22) |                         // pad_interval: 32 DWORDs
                (3u << 25));                         // pad_amount: 4 DWORDs
  g1[1] = (int)((cols & 0xffffu) << 16);             // tensor_dim0[15:0]
  g1[2] = (int)(((cols >> 16) & 0xffffu) | ((rows & 0xffffu) << 16));
  g1[3] = (int)(((rows >> 16) & 0xffffu) | ((cols & 0xffffu) << 16)); // tile_dim0
  g1[4] = (int)(rows & 0xffffu);                     // tile_dim1, tile_dim2=0
  g1[5] = (int)stride_elems;                         // tensor_dim0_stride lo
  g1[6] = 0;
  g1[7] = 0;
  i32x4 gz = {0, 0, 0, 0};
#if __clang_major__ >= 23
  i32x8 gz8 = {0, 0, 0, 0, 0, 0, 0, 0};
  __builtin_amdgcn_tensor_load_to_lds(g0, g1, gz, gz, gz8, 0);
#else
  __builtin_amdgcn_tensor_load_to_lds(g0, g1, gz, gz, 0);
#endif
}

// ---- GEMM: C[b][M,1024] = A[b][M,1024] * Bw[1024,1024] + bias --------------
// Block tile 128x128, 8 waves -> each wave 32x64 (2x4 accum tiles of 16x16).
template <typename TA, typename TC>
__global__ __launch_bounds__(256) void gemm1024_bias(
    const TA* __restrict__ A, const float* __restrict__ Bw,
    const float* __restrict__ bias, TC* __restrict__ C,
    long long strideA, long long strideC) {
  constexpr int NK = 1024;
  constexpr int TK = 64;
  constexpr int LD = TK + 8;
  __shared__ bf16 Alds[128 * LD];
  __shared__ bf16 Blds[128 * LD];

  const int tid = threadIdx.x;
  const int wave = tid >> 5;
  const int wm = wave & 3, wn = wave >> 2;
  const long long m0 = (long long)blockIdx.y * 128;
  const int n0 = blockIdx.x * 128;
  const TA* Ab = A + (long long)blockIdx.z * strideA + m0 * NK;
  TC* Cb = C + (long long)blockIdx.z * strideC + m0 * NK + n0;

  const int ac4 = tid & 15, ar = tid >> 4;
  const int bc4 = tid & 31, br = tid >> 5;

  bf16x4 aReg[8];
  float4 bReg[8];
  auto loadG = [&](int kt) {
    const TA* ap = Ab + kt * TK + ac4 * 4 + (long long)ar * NK;
#pragma unroll
    for (int p = 0; p < 8; ++p) aReg[p] = ld4bf(ap + (long long)p * 16 * NK);
    const float* bp = Bw + (long long)(kt * TK + br) * NK + n0 + bc4 * 4;
#pragma unroll
    for (int p = 0; p < 8; ++p)
      bReg[p] = *(const float4*)(bp + (long long)p * 8 * NK);
  };
  auto stageL = [&]() {
#pragma unroll
    for (int p = 0; p < 8; ++p)
      *(bf16x4*)&Alds[(ar + p * 16) * LD + ac4 * 4] = aReg[p];
#pragma unroll
    for (int p = 0; p < 8; ++p) {  // transpose B into [n][k]
      const int k = br + p * 8, n = bc4 * 4;
      Blds[(n + 0) * LD + k] = (bf16)bReg[p].x;
      Blds[(n + 1) * LD + k] = (bf16)bReg[p].y;
      Blds[(n + 2) * LD + k] = (bf16)bReg[p].z;
      Blds[(n + 3) * LD + k] = (bf16)bReg[p].w;
    }
  };

  const f32x8 z8 = {0, 0, 0, 0, 0, 0, 0, 0};
  f32x8 acc[2][4];
#pragma unroll
  for (int mi = 0; mi < 2; ++mi)
#pragma unroll
    for (int t = 0; t < 4; ++t) acc[mi][t] = z8;

  loadG(0);
  constexpr int NKT = NK / TK;  // 16
  for (int kt = 0; kt < NKT; ++kt) {
    stageL();
    __syncthreads();
    if (kt + 1 < NKT) loadG(kt + 1);
#pragma unroll
    for (int kk = 0; kk < TK; kk += 32) {
      bf16x16 a0 = frag_a_ld(&Alds[(wm * 32) * LD + kk], LD);
      bf16x16 a1 = frag_a_ld(&Alds[(wm * 32 + 16) * LD + kk], LD);
#pragma unroll
      for (int t = 0; t < 4; ++t) {
        bf16x16 bt = frag_b_ld(&Blds[(wn * 64 + t * 16) * LD + kk], LD);
        acc[0][t] = wmma_bf16(a0, bt, acc[0][t]);
        acc[1][t] = wmma_bf16(a1, bt, acc[1][t]);
      }
    }
    __syncthreads();
  }

  const int lane = tid & 31, nlo = lane & 15, hh = lane >> 4;
#pragma unroll
  for (int mi = 0; mi < 2; ++mi) {
#pragma unroll
    for (int t = 0; t < 4; ++t) {
      const int col = wn * 64 + t * 16 + nlo;
      const float bb = bias[n0 + col];
      TC* cp = Cb + (long long)(wm * 32 + mi * 16 + hh * 8) * NK + col;
#pragma unroll
      for (int r = 0; r < 8; ++r) st_c(cp + (long long)r * NK, acc[mi][t][r] + bb);
    }
  }
}

// ---- Fused flash attention (bf16 in/out, f32 softmax) ----------------------
// Q fragments live in registers (staged once via TDM through the Ps region).
// K chunks: TDM double-buffered (DMA for chunk i+1 overlaps compute of i).
// V chunks: manual transposed staging, double-buffered.
// One barrier per chunk.
__global__ __launch_bounds__(256) void attn_fused(
    const bf16* __restrict__ qb, const bf16* __restrict__ kb,
    const bf16* __restrict__ vb, const int* __restrict__ mask,
    bf16* __restrict__ ctx) {
  constexpr int QL = 1024, KV = 1280, H = 16, HD = 64, CH = 64;
  constexpr int NCH = KV / CH;  // 20
  constexpr int LD = 72;
  __shared__ bf16 Ks[2][64 * LD];  // [kv][hd]  (B layout)  -- TDM filled
  __shared__ bf16 Vs[2][64 * LD];  // [hd][kv]  (transposed, manual)
  __shared__ bf16 Ps[128 * LD];    // Q staging (pre-loop) then P (per-wave rows)

  const int tid = threadIdx.x, wave = tid >> 5;
  const int b = blockIdx.z, h = blockIdx.y, q0 = blockIdx.x * 128;
  const int lane = tid & 31, nlo = lane & 15, hh = lane >> 4;
  const f32x8 z8 = {0, 0, 0, 0, 0, 0, 0, 0};

  const bf16* qtile = qb + (((long long)b * QL + q0) * H + h) * HD;
  const bf16* kbase = kb + ((long long)b * KV * H + h) * HD;
  if (wave == 0) {
    tdm_load_2d_bf16((unsigned)(uintptr_t)Ps, qtile, 128u, 64u, (unsigned)(H * HD));
    tdm_load_2d_bf16((unsigned)(uintptr_t)&Ks[0][0], kbase, 64u, 64u,
                     (unsigned)(H * HD));
    __builtin_amdgcn_s_wait_tensorcnt(1);  // in-order: Q tile complete
  }
  __syncthreads();

  // Hoist Q A-fragments into registers (16 VGPRs per wave); the wave will
  // later overwrite only its own Ps rows -> same-wave LDS is in-order.
  const bf16x16 aq0 = frag_a_ld(&Ps[(wave * 16) * LD + 0], LD);
  const bf16x16 aq1 = frag_a_ld(&Ps[(wave * 16) * LD + 32], LD);

  f32x8 o[4];
#pragma unroll
  for (int t = 0; t < 4; ++t) o[t] = z8;
  float m_run[8], l_run[8];
#pragma unroll
  for (int r = 0; r < 8; ++r) { m_run[r] = -3.0e38f; l_run[r] = 0.0f; }

  const int* mbase = mask + ((long long)b * QL + q0 + wave * 16) * KV;

  for (int i = 0; i < NCH; ++i) {
    const int kv0 = i * CH, bi = i & 1;

    {  // V chunk staged manually with transpose into [hd][kv]
      const int c4 = tid & 15, r0 = tid >> 4;
#pragma unroll
      for (int p = 0; p < 4; ++p) {
        int row = r0 + p * 16;
        const bf16* src = vb + (((long long)b * KV + kv0 + row) * H + h) * HD + c4 * 4;
        bf16x4 v = *(const bf16x4*)src;
        Vs[bi][(c4 * 4 + 0) * LD + row] = v[0];
        Vs[bi][(c4 * 4 + 1) * LD + row] = v[1];
        Vs[bi][(c4 * 4 + 2) * LD + row] = v[2];
        Vs[bi][(c4 * 4 + 3) * LD + row] = v[3];
      }
    }
    if (wave == 0) __builtin_amdgcn_s_wait_tensorcnt(0);  // K chunk i landed
    __syncthreads();  // barrier_i: K[i&1]/V[i&1] ready, compute_{i-1} retired

    // prefetch next K chunk via TDM, overlapped with this chunk's compute
    if (wave == 0 && i + 1 < NCH)
      tdm_load_2d_bf16((unsigned)(uintptr_t)&Ks[(i + 1) & 1][0],
                       kbase + (long long)(i + 1) * CH * H * HD, 64u, 64u,
                       (unsigned)(H * HD));

    // S = Q * K^T  (16 q x 64 kv per wave)
    f32x8 s[4];
#pragma unroll
    for (int t = 0; t < 4; ++t) s[t] = z8;
#pragma unroll
    for (int kk = 0; kk < HD; kk += 32) {
      const bf16x16 aq = kk ? aq1 : aq0;
#pragma unroll
      for (int t = 0; t < 4; ++t) {
        bf16x16 bk_ = frag_b_ld(&Ks[bi][(t * 16) * LD + kk], LD);
        s[t] = wmma_bf16(aq, bk_, s[t]);
      }
    }

    // mask bias: where(mask>0, 0, -1e10)
#pragma unroll
    for (int t = 0; t < 4; ++t)
#pragma unroll
      for (int r = 0; r < 8; ++r) {
        int mrow = hh * 8 + r;
        int mv = mbase[(long long)mrow * KV + kv0 + t * 16 + nlo];
        if (mv <= 0) s[t][r] += -1.0e10f;
      }

    // online softmax: each row lives in 16 lanes of one half-wave
    float pm[8];
#pragma unroll
    for (int r = 0; r < 8; ++r)
      pm[r] = fmaxf(fmaxf(s[0][r], s[1][r]), fmaxf(s[2][r], s[3][r]));
#pragma unroll
    for (int off = 8; off >= 1; off >>= 1)
#pragma unroll
      for (int r = 0; r < 8; ++r) pm[r] = fmaxf(pm[r], __shfl_xor(pm[r], off, 32));

    float rs[8];
#pragma unroll
    for (int r = 0; r < 8; ++r) {
      float nm = fmaxf(m_run[r], pm[r]);
      float sc = __expf(m_run[r] - nm);
      m_run[r] = nm;
      float a = 0.0f;
#pragma unroll
      for (int t = 0; t < 4; ++t) {
        float p = __expf(s[t][r] - nm);
        s[t][r] = p;
        a += p;
      }
      rs[r] = a;
      l_run[r] *= sc;
#pragma unroll
      for (int t = 0; t < 4; ++t) o[t][r] *= sc;
    }
#pragma unroll
    for (int off = 8; off >= 1; off >>= 1)
#pragma unroll
      for (int r = 0; r < 8; ++r) rs[r] += __shfl_xor(rs[r], off, 32);
#pragma unroll
    for (int r = 0; r < 8; ++r) l_run[r] += rs[r];

    // redistribute P (C layout -> A layout) via this wave's private Ps rows
#pragma unroll
    for (int t = 0; t < 4; ++t)
#pragma unroll
      for (int r = 0; r < 8; ++r)
        Ps[(wave * 16 + hh * 8 + r) * LD + t * 16 + nlo] = (bf16)s[t][r];

    // O += P * V  (same-wave LDS is in-order; no barrier needed for Ps)
#pragma unroll
    for (int kk = 0; kk < CH; kk += 32) {
      bf16x16 ap = frag_a_ld(&Ps[(wave * 16) * LD + kk], LD);
#pragma unroll
      for (int t = 0; t < 4; ++t) {
        bf16x16 bv_ = frag_b_ld(&Vs[bi][(t * 16) * LD + kk], LD);
        o[t] = wmma_bf16(ap, bv_, o[t]);
      }
    }
    // no trailing barrier: buffer reuse is fenced by barrier_{i+1}
  }

#pragma unroll
  for (int r = 0; r < 8; ++r) l_run[r] = 1.0f / l_run[r];
#pragma unroll
  for (int t = 0; t < 4; ++t)
#pragma unroll
    for (int r = 0; r < 8; ++r) {
      long long q = q0 + wave * 16 + hh * 8 + r;
      ctx[(((long long)b * QL + q) * H + h) * HD + t * 16 + nlo] =
          (bf16)(o[t][r] * l_run[r]);
    }
}

// ---- host ------------------------------------------------------------------
extern "C" void kernel_launch(void* const* d_in, const int* in_sizes, int n_in,
                              void* d_out, int out_size, void* d_ws, size_t ws_size,
                              hipStream_t stream) {
  const float* inq  = (const float*)d_in[0];
  const float* kvs  = (const float*)d_in[1];
  const float* kvc  = (const float*)d_in[2];
  const int*   mask = (const int*)d_in[3];
  const float* Wq   = (const float*)d_in[4];
  const float* bq   = (const float*)d_in[5];
  const float* Wk   = (const float*)d_in[6];
  const float* bk   = (const float*)d_in[7];
  const float* Wv   = (const float*)d_in[8];
  const float* bv   = (const float*)d_in[9];
  const float* Wkc  = (const float*)d_in[10];
  const float* bkc  = (const float*)d_in[11];
  const float* Wvc  = (const float*)d_in[12];
  const float* bvc  = (const float*)d_in[13];
  const float* Wo   = (const float*)d_in[14];
  const float* bo   = (const float*)d_in[15];
  float* out = (float*)d_out;

  // bf16 intermediates: q [4,1024,16,64], k/v [4,1280,16,64], ctx [4,1024,16,64]
  bf16* qb   = (bf16*)d_ws;
  bf16* kbuf = qb + (size_t)4 * 1024 * 1024;
  bf16* vbuf = kbuf + (size_t)4 * 1280 * 1024;
  bf16* ctx  = vbuf + (size_t)4 * 1280 * 1024;

  dim3 blk(256);
  const long long SA = 1024LL * 1024, SAc = 256LL * 1024;
  const long long SC = 1024LL * 1024, SCkv = 1280LL * 1024;

  gemm1024_bias<float, bf16><<<dim3(8, 8, 4), blk, 0, stream>>>(inq, Wq, bq, qb, SA, SC);
  gemm1024_bias<float, bf16><<<dim3(8, 8, 4), blk, 0, stream>>>(kvs, Wk, bk, kbuf, SA, SCkv);
  gemm1024_bias<float, bf16><<<dim3(8, 8, 4), blk, 0, stream>>>(kvs, Wv, bv, vbuf, SA, SCkv);
  gemm1024_bias<float, bf16><<<dim3(8, 2, 4), blk, 0, stream>>>(
      kvc, Wkc, bkc, kbuf + 1024LL * 1024, SAc, SCkv);
  gemm1024_bias<float, bf16><<<dim3(8, 2, 4), blk, 0, stream>>>(
      kvc, Wvc, bvc, vbuf + 1024LL * 1024, SAc, SCkv);
  attn_fused<<<dim3(8, 16, 4), blk, 0, stream>>>(qb, kbuf, vbuf, mask, ctx);
  gemm1024_bias<bf16, float><<<dim3(8, 8, 4), blk, 0, stream>>>(ctx, Wo, bo, out, SA, SC);
}
